// GroupedQueryAttention_56959856280094
// MI455X (gfx1250) — matrix operand order repaired and checked by reading, not compiled, not run on hardware
//
#include <hip/hip_runtime.h>
#include <cmath>

// ---------------- problem constants ----------------
#define B_     4
#define T_     512
#define DIN    2048
#define H_     16
#define G_     4
#define HD_    128
#define CACHE_ 3584
#define S_     4096   // CACHE_ + T_

#define KST    144    // LDS K row stride (128 data + 16 pad halves) = 288B, 32B-aligned
#define VST    48     // LDS V row stride (32 data + 16 pad halves)  =  96B, 32B-aligned

typedef __attribute__((ext_vector_type(16))) _Float16 v16h;
typedef __attribute__((ext_vector_type(8)))  _Float16 v8h;
typedef __attribute__((ext_vector_type(8)))  float    v8f;

// ---------------- WMMA helpers (CDNA5, wave32) ----------------
__device__ __forceinline__ v8f wmma16(v16h a, v16h b, v8f c) {
  return __builtin_amdgcn_wmma_f32_16x16x32_f16(
      false, a, false, b, (short)0, c, false, false);
}

// A-fragment 16x32 f16 from row-major tile (row stride ld halves).
// lanes 0-15 -> M=lane, K={0..7,16..23}; lanes 16-31 -> K={8..15,24..31}.
__device__ __forceinline__ v16h load_frag_a(const _Float16* base, int ld) {
  const int lane = threadIdx.x & 31;
  const int row  = lane & 15;
  const int grp  = lane >> 4;
  const _Float16* p = base + (size_t)row * ld + grp * 8;
  v8h lo = *(const v8h*)(p);
  v8h hi = *(const v8h*)(p + 16);
  v16h r;
#pragma unroll
  for (int i = 0; i < 8; ++i) { r[i] = lo[i]; r[8 + i] = hi[i]; }
  return r;
}

// B-fragment 32x16 f16 where B[k][n] = base[n*ld + k] (source N-major).
// lanes 0-15 -> N=lane, K={0..7,16..23}; lanes 16-31 -> N=lane-16, K={8..15,24..31}.
__device__ __forceinline__ v16h load_frag_bT(const _Float16* base, int ld) {
  const int lane = threadIdx.x & 31;
  const int n    = lane & 15;
  const int grp  = lane >> 4;
  const _Float16* p = base + (size_t)n * ld + grp * 8;
  v8h lo = *(const v8h*)(p);
  v8h hi = *(const v8h*)(p + 16);
  v16h r;
#pragma unroll
  for (int i = 0; i < 8; ++i) { r[i] = lo[i]; r[8 + i] = hi[i]; }
  return r;
}

__device__ __forceinline__ float hmax16(float x) {
  x = fmaxf(x, __shfl_xor(x, 1, 32));
  x = fmaxf(x, __shfl_xor(x, 2, 32));
  x = fmaxf(x, __shfl_xor(x, 4, 32));
  x = fmaxf(x, __shfl_xor(x, 8, 32));
  return x;
}
__device__ __forceinline__ float hsum16(float x) {
  x += __shfl_xor(x, 1, 32);
  x += __shfl_xor(x, 2, 32);
  x += __shfl_xor(x, 4, 32);
  x += __shfl_xor(x, 8, 32);
  return x;
}

// ---------------- CDNA5 async global -> LDS (ASYNCcnt path) ----------------
__device__ __forceinline__ void async_g2l_b128(const _Float16* gp, _Float16* lp) {
  unsigned l = (unsigned)(size_t)lp;     // LDS aperture: addr[31:0] = LDS offset
  asm volatile("global_load_async_to_lds_b128 %0, %1, off"
               :: "v"(l), "v"(gp) : "memory");
}
__device__ __forceinline__ void wait_async0() {
  asm volatile("s_wait_asynccnt 0" ::: "memory");
}

// ---------------- elementwise prep kernels ----------------
__global__ void k_f32_to_f16(const float* __restrict__ src,
                             _Float16* __restrict__ dst, int n) {
  int i = blockIdx.x * blockDim.x + threadIdx.x;
  if (i < n) dst[i] = (_Float16)src[i];
}

// wt[n*K + k] = w[k*N + n]  (f16, N-major so B-frags are two 16B loads along k)
__global__ void k_transpose_w(const float* __restrict__ w,
                              _Float16* __restrict__ wt, int K, int N) {
  int i = blockIdx.x * blockDim.x + threadIdx.x;
  if (i >= K * N) return;
  int k = i / N, n = i % N;
  wt[(size_t)n * K + k] = (_Float16)w[i];
}

// KV cache -> f32 outputs + f16 staging (V transposed to [hd][s]).
__global__ void k_cache_prep(const float* __restrict__ ck, const float* __restrict__ cv,
                             float* __restrict__ out_k, float* __restrict__ out_v,
                             _Float16* __restrict__ kf, _Float16* __restrict__ vt, int n) {
  int i = blockIdx.x * blockDim.x + threadIdx.x;
  if (i >= n) return;                 // n = B*G*CACHE*HD
  int d  = i & (HD_ - 1);
  int s  = (i >> 7) % CACHE_;
  int bg = i / (CACHE_ * HD_);
  size_t o = ((size_t)bg * S_ + s) * HD_ + d;
  float kv = ck[i], vv = cv[i];
  out_k[o] = kv;  kf[o] = (_Float16)kv;
  out_v[o] = vv;  vt[((size_t)bg * HD_ + d) * S_ + s] = (_Float16)vv;
}

// ---------------- fused QKV projection + RMSNorm + RoPE ----------------
// grid.x = B*T/32 (two 16-row sub-tiles per wave -> each B-frag feeds 2 WMMAs),
// grid.y = 24 heads (0..15 q, 16..19 k, 20..23 v). One wave per block.
__global__ __launch_bounds__(32)
void k_proj(const _Float16* __restrict__ x16, const _Float16* __restrict__ wt,
            const float* __restrict__ cosT, const float* __restrict__ sinT,
            const int* __restrict__ pos_ids,
            const float* __restrict__ qn_w, const float* __restrict__ kn_w,
            _Float16* __restrict__ qf, _Float16* __restrict__ kf,
            _Float16* __restrict__ vt,
            float* __restrict__ out_k, float* __restrict__ out_v) {
  const int r0  = blockIdx.x * 32;       // global row base over B*T
  const int hh  = blockIdx.y;
  const int lane = threadIdx.x, n15 = lane & 15, grp = lane >> 4;

  const _Float16* xb = x16 + (size_t)r0 * DIN;
  const _Float16* wb = wt + (size_t)hh * HD_ * DIN;

  v8f acc[2][8];
#pragma unroll
  for (int rr = 0; rr < 2; ++rr)
#pragma unroll
    for (int t = 0; t < 8; ++t) acc[rr][t] = (v8f)(0.0f);

  for (int k0 = 0; k0 < DIN; k0 += 32) {
    v16h a0 = load_frag_a(xb + k0, DIN);
    v16h a1 = load_frag_a(xb + (size_t)16 * DIN + k0, DIN);
#pragma unroll
    for (int t = 0; t < 8; ++t) {
      v16h b = load_frag_bT(wb + (size_t)(t * 16) * DIN + k0, DIN);
      acc[0][t] = wmma16(a0, b, acc[0][t]);
      acc[1][t] = wmma16(a1, b, acc[1][t]);
    }
  }

  const bool is_q = hh < 16, is_k = (hh >= 16 && hh < 20);
#pragma unroll
  for (int rr = 0; rr < 2; ++rr) {
    int rbase = r0 + rr * 16;
    if (is_q || is_k) {
      // per-row RMSNorm over HD=128
      const float* nw = is_q ? qn_w : kn_w;
      float rs[8];
#pragma unroll
      for (int r = 0; r < 8; ++r) {
        float s = 0.f;
#pragma unroll
        for (int t = 0; t < 8; ++t) s += acc[rr][t][r] * acc[rr][t][r];
        rs[r] = rsqrtf(hsum16(s) * (1.f / HD_) + 1e-6f);
      }
#pragma unroll
      for (int t = 0; t < 8; ++t) {
        float w = nw[t * 16 + n15];
#pragma unroll
        for (int r = 0; r < 8; ++r) acc[rr][t][r] = acc[rr][t][r] * rs[r] * w;
      }
      // RoPE: col pair (i, i+64) -> fragment pair (t, t+4)
      int pos[8];
#pragma unroll
      for (int r = 0; r < 8; ++r) pos[r] = pos_ids[rbase + grp * 8 + r];
      v8f na[8];
#pragma unroll
      for (int t = 0; t < 8; ++t) {
        int col = t * 16 + n15;
#pragma unroll
        for (int r = 0; r < 8; ++r) {
          float c   = cosT[(size_t)pos[r] * HD_ + col];
          float s   = sinT[(size_t)pos[r] * HD_ + col];
          float rot = (t < 4) ? -acc[rr][t + 4][r] : acc[rr][t - 4][r];
          na[t][r]  = acc[rr][t][r] * c + rot * s;
        }
      }
#pragma unroll
      for (int t = 0; t < 8; ++t) acc[rr][t] = na[t];
    }
    // scatter stores from C layout
#pragma unroll
    for (int r = 0; r < 8; ++r) {
      int rowg = rbase + grp * 8 + r;
      int b = rowg / T_, tt = rowg % T_;
      if (is_q) {
        _Float16* dst = qf + (((size_t)(b * H_ + hh) * T_) + tt) * HD_;
#pragma unroll
        for (int t = 0; t < 8; ++t) dst[t * 16 + n15] = (_Float16)acc[rr][t][r];
      } else if (is_k) {
        int g = hh - 16;
        size_t base = (((size_t)(b * G_ + g) * S_) + (CACHE_ + tt)) * HD_;
#pragma unroll
        for (int t = 0; t < 8; ++t) {
          float v = acc[rr][t][r];
          kf[base + t * 16 + n15]    = (_Float16)v;
          out_k[base + t * 16 + n15] = v;
        }
      } else {
        int g = hh - 20, sidx = CACHE_ + tt;
        size_t base = (((size_t)(b * G_ + g) * S_) + sidx) * HD_;
        size_t vtb  = ((size_t)(b * G_ + g) * HD_) * S_;
#pragma unroll
        for (int t = 0; t < 8; ++t) {
          float v = acc[rr][t][r];
          out_v[base + t * 16 + n15]                   = v;
          vt[vtb + (size_t)(t * 16 + n15) * S_ + sidx] = (_Float16)v;
        }
      }
    }
  }
}

// ---------------- flash attention ----------------
// Block = 256 threads = 8 waves sharing one (b,g): 4 q-heads x 2 q-tiles.
// K/V chunks (32 keys) staged in LDS via async global->LDS, double-buffered,
// so K/V is read from L2 once per block instead of once per wave (8x less).
__global__ __launch_bounds__(256)
void k_attn(const _Float16* __restrict__ qf, const _Float16* __restrict__ kf,
            const _Float16* __restrict__ vt, const unsigned char* __restrict__ mask,
            _Float16* __restrict__ ctx16) {
  __shared__ __align__(128) _Float16 sK[2][32 * KST];    // 2 x 9.0 KB
  __shared__ __align__(128) _Float16 sV[2][128 * VST];   // 2 x 12.0 KB
  __shared__ __align__(128) _Float16 sP[8][16 * 32];     // 8 KB (per-wave P tiles)

  const int tid = threadIdx.x, w = tid >> 5;
  const int lane = tid & 31, n15 = lane & 15, grp = lane >> 4;
  const int g = blockIdx.y, b = blockIdx.z;
  const int h  = g * (H_ / G_) + (w & 3);
  const int t0 = blockIdx.x * 32 + (w >> 2) * 16;
  const float scale = 0.08838834764831845f;   // 1/sqrt(128)

  const _Float16* qb = qf + (((size_t)(b * H_ + h) * T_) + t0) * HD_;
  const _Float16* kb = kf + ((size_t)(b * G_ + g) * S_) * HD_;
  const _Float16* vb = vt + ((size_t)(b * G_ + g) * HD_) * S_;
  const unsigned char* mb = mask + ((size_t)(b * T_ + t0)) * S_;

  v16h qa[4];
#pragma unroll
  for (int kk = 0; kk < 4; ++kk) qa[kk] = load_frag_a(qb + kk * 32, HD_);

  v8f acc[8];
#pragma unroll
  for (int t = 0; t < 8; ++t) acc[t] = (v8f)(0.0f);
  float M[8], L[8];
#pragma unroll
  for (int r = 0; r < 8; ++r) { M[r] = -INFINITY; L[r] = 0.f; }

  // cooperative async stage of one 32-key chunk: K 32x128 (row-major),
  // V 128x32 (from transposed V, row = hd). 1024 b128 transfers / 256 threads.
  auto stage = [&](int s0, int buf) {
    const _Float16* ks = kb + (size_t)s0 * HD_;
    _Float16* dK = &sK[buf][0];
    _Float16* dV = &sV[buf][0];
#pragma unroll
    for (int i = 0; i < 2; ++i) {
      int idx = i * 256 + tid;                 // 0..511
      int row = idx >> 4, c = (idx & 15) * 8;
      async_g2l_b128(ks + (size_t)row * HD_ + c, dK + row * KST + c);
    }
#pragma unroll
    for (int i = 0; i < 2; ++i) {
      int idx = i * 256 + tid;                 // 0..511
      int d = idx >> 2, c = (idx & 3) * 8;
      async_g2l_b128(vb + (size_t)d * S_ + s0 + c, dV + d * VST + c);
    }
  };

  stage(0, 0);
  wait_async0();
  __syncthreads();

  const int nIter = S_ / 32;
  for (int it = 0; it < nIter; ++it) {
    const int s0 = it * 32, cur = it & 1;
    if (it + 1 < nIter) stage(s0 + 32, cur ^ 1);   // prefetch next chunk

    // scores: 16 rows x 32 keys, contraction over HD in 4 chunks of 32
    v8f sf0 = (v8f)(0.0f), sf1 = (v8f)(0.0f);
    const _Float16* Kc = &sK[cur][0];
#pragma unroll
    for (int kk = 0; kk < 4; ++kk) {
      v16h b0 = load_frag_bT(Kc + kk * 32, KST);
      v16h b1 = load_frag_bT(Kc + 16 * KST + kk * 32, KST);
      sf0 = wmma16(qa[kk], b0, sf0);
      sf1 = wmma16(qa[kk], b1, sf1);
    }
    // scale + mask + online softmax
    float corr[8];
#pragma unroll
    for (int r = 0; r < 8; ++r) {
      int m = grp * 8 + r;
      float x0 = sf0[r] * scale, x1 = sf1[r] * scale;
      if (mb[(size_t)m * S_ + s0 + n15])      x0 = -INFINITY;
      if (mb[(size_t)m * S_ + s0 + 16 + n15]) x1 = -INFINITY;
      float tm = hmax16(fmaxf(x0, x1));
      float nm = fmaxf(fmaxf(M[r], tm), -3.0e38f);
      float c_ = __expf(M[r] - nm);
      float p0 = __expf(x0 - nm), p1 = __expf(x1 - nm);
      L[r] = L[r] * c_ + hsum16(p0 + p1);
      M[r] = nm; corr[r] = c_;
      sf0[r] = p0; sf1[r] = p1;
    }
#pragma unroll
    for (int t = 0; t < 8; ++t)
#pragma unroll
      for (int r = 0; r < 8; ++r) acc[t][r] *= corr[r];

    // P (two C-frags) -> per-wave LDS tile -> A-fragment (same-wave, DScnt only)
    _Float16* Pw = &sP[w][0];
#pragma unroll
    for (int r = 0; r < 8; ++r) {
      int m = grp * 8 + r;
      Pw[m * 32 + n15]      = (_Float16)sf0[r];
      Pw[m * 32 + 16 + n15] = (_Float16)sf1[r];
    }
    asm volatile("s_wait_dscnt 0" ::: "memory");
    v16h pa = load_frag_a(Pw, 32);

    // ctx += P(16x32) @ V(32x128) from LDS (V rows = hd)
    const _Float16* Vc = &sV[cur][0];
#pragma unroll
    for (int t = 0; t < 8; ++t) {
      v16h bv = load_frag_bT(Vc + (t * 16) * VST, VST);
      acc[t] = wmma16(pa, bv, acc[t]);
    }

    wait_async0();        // next chunk landed in LDS
    __syncthreads();      // all waves done with current buffers
  }

  // normalize and store ctx as f16 [b, t, h*HD + d]
#pragma unroll
  for (int r = 0; r < 8; ++r) {
    float inv = (L[r] > 0.f) ? 1.f / L[r] : 0.f;
    int tg = t0 + grp * 8 + r;
    _Float16* dst = ctx16 + ((size_t)(b * T_ + tg)) * (H_ * HD_) + h * HD_;
#pragma unroll
    for (int t = 0; t < 8; ++t) dst[t * 16 + n15] = (_Float16)(acc[t][r] * inv);
  }
}

// ---------------- output projection: ctx @ o_w -> f32 ----------------
__global__ __launch_bounds__(32)
void k_oproj(const _Float16* __restrict__ ctx16, const _Float16* __restrict__ owt,
             float* __restrict__ out) {
  const int r0 = blockIdx.x * 32, ct = blockIdx.y;
  const int lane = threadIdx.x, n15 = lane & 15, grp = lane >> 4;
  const _Float16* xb = ctx16 + (size_t)r0 * (H_ * HD_);
  const _Float16* wb = owt + (size_t)(ct * 128) * (H_ * HD_);
  v8f acc[2][8];
#pragma unroll
  for (int rr = 0; rr < 2; ++rr)
#pragma unroll
    for (int t = 0; t < 8; ++t) acc[rr][t] = (v8f)(0.0f);
  for (int k0 = 0; k0 < H_ * HD_; k0 += 32) {
    v16h a0 = load_frag_a(xb + k0, H_ * HD_);
    v16h a1 = load_frag_a(xb + (size_t)16 * (H_ * HD_) + k0, H_ * HD_);
#pragma unroll
    for (int t = 0; t < 8; ++t) {
      v16h b = load_frag_bT(wb + (size_t)(t * 16) * (H_ * HD_) + k0, H_ * HD_);
      acc[0][t] = wmma16(a0, b, acc[0][t]);
      acc[1][t] = wmma16(a1, b, acc[1][t]);
    }
  }
#pragma unroll
  for (int rr = 0; rr < 2; ++rr)
#pragma unroll
    for (int r = 0; r < 8; ++r) {
      int m = grp * 8 + r;
      float* dst = out + ((size_t)(r0 + rr * 16 + m)) * DIN + ct * 128;
#pragma unroll
      for (int t = 0; t < 8; ++t) dst[t * 16 + n15] = acc[rr][t][r];
    }
}

// ---------------- host launcher ----------------
extern "C" void kernel_launch(void* const* d_in, const int* in_sizes, int n_in,
                              void* d_out, int out_size, void* d_ws, size_t ws_size,
                              hipStream_t stream) {
  (void)in_sizes; (void)n_in; (void)out_size; (void)ws_size;
  const float* x        = (const float*)d_in[0];
  const unsigned char* mask = (const unsigned char*)d_in[1];   // jnp bool = 1 byte
  const float* cosT     = (const float*)d_in[2];
  const float* sinT     = (const float*)d_in[3];
  const int*   pos_ids  = (const int*)d_in[4];
  const float* cache_k  = (const float*)d_in[5];
  const float* cache_v  = (const float*)d_in[6];
  const float* q_w      = (const float*)d_in[7];
  const float* k_w      = (const float*)d_in[8];
  const float* v_w      = (const float*)d_in[9];
  const float* o_w      = (const float*)d_in[10];
  const float* qn_w     = (const float*)d_in[11];
  const float* kn_w     = (const float*)d_in[12];

  float* out_ctx = (float*)d_out;                              // [B,T,DIN]
  float* out_k   = out_ctx + (size_t)B_ * T_ * DIN;            // [B,G,S,HD]
  float* out_v   = out_k + (size_t)B_ * G_ * S_ * HD_;         // [B,G,S,HD]

  char* wsb = (char*)d_ws;
  size_t off = 0;
  auto take = [&](size_t bytes) {
    char* p = wsb + off;
    off += (bytes + 255) & ~(size_t)255;
    return p;
  };
  _Float16* x16    = (_Float16*)take((size_t)B_ * T_ * DIN * 2);          // 8 MB
  _Float16* wt_all = (_Float16*)take((size_t)3072 * DIN * 2);             // 12 MB (q|k|v)
  _Float16* owt    = (_Float16*)take((size_t)DIN * (H_ * HD_) * 2);       // 8 MB
  _Float16* qf     = (_Float16*)take((size_t)B_ * H_ * T_ * HD_ * 2);     // 8 MB
  _Float16* kf     = (_Float16*)take((size_t)B_ * G_ * S_ * HD_ * 2);     // 16 MB
  _Float16* vt     = (_Float16*)take((size_t)B_ * G_ * HD_ * S_ * 2);     // 16 MB
  _Float16* ctx16  = (_Float16*)take((size_t)B_ * T_ * H_ * HD_ * 2);     // 8 MB

  int n = B_ * T_ * DIN;
  k_f32_to_f16<<<(n + 255) / 256, 256, 0, stream>>>(x, x16, n);
  k_transpose_w<<<(DIN * 2048 + 255) / 256, 256, 0, stream>>>(q_w, wt_all, DIN, 2048);
  k_transpose_w<<<(DIN * 512 + 255) / 256, 256, 0, stream>>>(k_w, wt_all + (size_t)2048 * DIN, DIN, 512);
  k_transpose_w<<<(DIN * 512 + 255) / 256, 256, 0, stream>>>(v_w, wt_all + (size_t)2560 * DIN, DIN, 512);
  k_transpose_w<<<(2048 * 2048 + 255) / 256, 256, 0, stream>>>(o_w, owt, H_ * HD_, DIN);

  n = B_ * G_ * CACHE_ * HD_;
  k_cache_prep<<<(n + 255) / 256, 256, 0, stream>>>(cache_k, cache_v, out_k, out_v, kf, vt, n);

  k_proj<<<dim3(B_ * T_ / 32, 24), 32, 0, stream>>>(
      x16, wt_all, cosT, sinT, pos_ids, qn_w, kn_w, qf, kf, vt, out_k, out_v);

  k_attn<<<dim3(T_ / 32, G_, B_), 256, 0, stream>>>(qf, kf, vt, mask, ctx16);

  k_oproj<<<dim3(B_ * T_ / 32, DIN / 128), 32, 0, stream>>>(ctx16, owt, out_ctx);
}
